// Memory_trans_update_39728447488513
// MI455X (gfx1250) — matrix operand order repaired and checked by reading, not compiled, not run on hardware
//
#include <hip/hip_runtime.h>
#include <hip/hip_bf16.h>

// ---------------------------------------------------------------------------
// Problem constants (reference: M=512 slots, D=256 dims, N=131072 queries)
// ---------------------------------------------------------------------------
#define MM 512
#define DD 256
#define NN 131072

#define BLOCK 256      // threads per block (8 waves of 32)
#define QTILE 128      // query rows per block in the score kernel
#define ROWQ  264      // padded LDS row (bf16 elements): 256 + 8 pad -> 528B rows
#define NCHUNK 128     // memory slots per chunk (4 chunks cover M=512)

typedef __attribute__((ext_vector_type(16))) __bf16 v16bf;
typedef __attribute__((ext_vector_type(8)))  float  v8f;

union ABFrag { v16bf v; uint4 q[2]; };

// ---------------------------------------------------------------------------
// bf16 helpers (round-to-nearest-even split: x ~= hi + lo)
// ---------------------------------------------------------------------------
__device__ __forceinline__ unsigned short f32_to_bf16(float x) {
  union { float f; unsigned int u; } c; c.f = x;
  unsigned int u = c.u;
  u += 0x7FFFu + ((u >> 16) & 1u);
  return (unsigned short)(u >> 16);
}
__device__ __forceinline__ float bf16_to_f32(unsigned short h) {
  union { unsigned int u; float f; } c; c.u = ((unsigned int)h) << 16;
  return c.f;
}

// float atomic max via signed/unsigned int trick (works for shared & global)
__device__ __forceinline__ void atomicMaxF(float* addr, float v) {
  if (v >= 0.0f) atomicMax((int*)addr, __float_as_int(v));
  else           atomicMin((unsigned int*)addr, __float_as_uint(v));
}

// ---------------------------------------------------------------------------
// Kernel 1: prep — split keys into bf16 hi/lo, zero accum, init colmax
// grid: 512 x 256 threads == M*D elements
// ---------------------------------------------------------------------------
__global__ __launch_bounds__(BLOCK) void prep_kernel(
    const float* __restrict__ keys,
    unsigned short* __restrict__ khi, unsigned short* __restrict__ klo,
    float* __restrict__ accum, float* __restrict__ colmax) {
  int e = blockIdx.x * BLOCK + threadIdx.x;   // 0 .. M*D-1
  float x = keys[e];
  unsigned short h = f32_to_bf16(x);
  float rem = x - bf16_to_f32(h);
  khi[e] = h;
  klo[e] = f32_to_bf16(rem);
  accum[e] = 0.0f;
  if (e < MM) colmax[e] = -__builtin_inff();
}

// ---------------------------------------------------------------------------
// Kernel 2: score — fused Q*K^T (bf16x3 WMMA) + row argmax/max + col max
// grid: N/QTILE = 1024 blocks x 256 threads, dynamic LDS ~137KB
// ---------------------------------------------------------------------------
__global__ __launch_bounds__(BLOCK) void score_kernel(
    const float* __restrict__ query,
    const unsigned short* __restrict__ khi,
    const unsigned short* __restrict__ klo,
    float* __restrict__ colmax,          // [M] global, atomic-max target
    float* __restrict__ rowmax,          // [N]
    int*   __restrict__ assign) {        // [N]
  extern __shared__ unsigned char smem[];
  unsigned short* qhi = (unsigned short*)smem;                  // QTILE*ROWQ
  unsigned short* qlo = qhi + QTILE * ROWQ;                     // QTILE*ROWQ
  float*          scm = (float*)(qlo + QTILE * ROWQ);           // [M]

  const int tid   = threadIdx.x;
  const int qbase = blockIdx.x * QTILE;

  // init block-local column max
  for (int c = tid; c < MM; c += BLOCK) scm[c] = -__builtin_inff();

  // stage query tile, split into bf16 hi/lo (coalesced: one row per iter)
  for (int row = 0; row < QTILE; ++row) {
    if (row + 8 < QTILE)
      __builtin_prefetch(&query[(size_t)(qbase + row + 8) * DD + tid], 0, 0);
    float x = query[(size_t)(qbase + row) * DD + tid];
    unsigned short h = f32_to_bf16(x);
    float rem = x - bf16_to_f32(h);
    qhi[row * ROWQ + tid] = h;
    qlo[row * ROWQ + tid] = f32_to_bf16(rem);
  }
  __syncthreads();

  const int wave  = tid >> 5;       // 0..7, each wave owns 16 query rows
  const int lane  = tid & 31;
  const int half  = lane >> 4;      // lane half selects K sub-range
  const int l15   = lane & 15;
  const int wrow0 = wave * 16;

  // running per-row best (rows half*8 + r), replicated across 16 lanes
  float rbest[8];
  int   rbestn[8];
#pragma unroll
  for (int r = 0; r < 8; ++r) { rbest[r] = -__builtin_inff(); rbestn[r] = 0; }

  for (int ch = 0; ch < MM / NCHUNK; ++ch) {
    const int slot0 = ch * NCHUNK;
    v8f acc[8];
#pragma unroll
    for (int nt = 0; nt < 8; ++nt) acc[nt] = (v8f){0,0,0,0,0,0,0,0};

#pragma unroll
    for (int ks = 0; ks < DD / 32; ++ks) {
      // --- A fragments (16x32 bf16): lane row = l15, K runs per ISA layout
      const int abase = (wrow0 + l15) * ROWQ + ks * 32 + half * 8;
      ABFrag a_h, a_l;
      a_h.q[0] = *(const uint4*)(qhi + abase);
      a_h.q[1] = *(const uint4*)(qhi + abase + 16);
      a_l.q[0] = *(const uint4*)(qlo + abase);
      a_l.q[1] = *(const uint4*)(qlo + abase + 16);

#pragma unroll
      for (int nt = 0; nt < 8; ++nt) {
        // --- B fragments (32x16 bf16): lane column = slot, K runs per ISA layout
        const int n = slot0 + nt * 16 + l15;
        const int bbase = n * DD + ks * 32 + half * 8;
        ABFrag b_h, b_l;
        b_h.q[0] = *(const uint4*)(khi + bbase);
        b_h.q[1] = *(const uint4*)(khi + bbase + 16);
        b_l.q[0] = *(const uint4*)(klo + bbase);
        b_l.q[1] = *(const uint4*)(klo + bbase + 16);

        // bf16x3 split product: qh*kh + qh*kl + ql*kh  (~f32-grade scores)
        acc[nt] = __builtin_amdgcn_wmma_f32_16x16x32_bf16(
            false, a_h.v, false, b_h.v, (short)0, acc[nt], false, false);
        acc[nt] = __builtin_amdgcn_wmma_f32_16x16x32_bf16(
            false, a_h.v, false, b_l.v, (short)0, acc[nt], false, false);
        acc[nt] = __builtin_amdgcn_wmma_f32_16x16x32_bf16(
            false, a_l.v, false, b_h.v, (short)0, acc[nt], false, false);
      }
    }

    // ---- row-wise argmax over this chunk (C layout: elem r -> row half*8+r,
    //      lane l15 -> column within 16-wide N tile)
#pragma unroll
    for (int r = 0; r < 8; ++r) {
      float bv = acc[0][r];
      int   bn = slot0 + l15;
#pragma unroll
      for (int nt = 1; nt < 8; ++nt) {
        float v = acc[nt][r];
        int   n = slot0 + nt * 16 + l15;
        if (v > bv) { bv = v; bn = n; }
      }
      // butterfly across the 16 lanes of this half (masks stay within half)
#pragma unroll
      for (int m = 8; m >= 1; m >>= 1) {
        float ov = __shfl_xor(bv, m, 32);
        int   on = __shfl_xor(bn, m, 32);
        if (ov > bv || (ov == bv && on < bn)) { bv = ov; bn = on; }
      }
      if (bv > rbest[r] || (bv == rbest[r] && bn < rbestn[r])) {
        rbest[r] = bv; rbestn[r] = bn;
      }
    }

    // ---- column max into block-local LDS array
#pragma unroll
    for (int nt = 0; nt < 8; ++nt) {
      float cm = acc[nt][0];
#pragma unroll
      for (int r = 1; r < 8; ++r) cm = fmaxf(cm, acc[nt][r]);
      cm = fmaxf(cm, __shfl_xor(cm, 16, 32));   // combine the two row-halves
      if (half == 0) atomicMaxF(&scm[slot0 + nt * 16 + l15], cm);
    }
  }

  // one writer lane per query row
  if (l15 < 8) {
    int m = half * 8 + l15;
    int q = qbase + wrow0 + m;
    rowmax[q] = rbest[l15];
    assign[q] = rbestn[l15];
  }

  __syncthreads();
  for (int c = tid; c < MM; c += BLOCK) atomicMaxF(&colmax[c], scm[c]);
}

// ---------------------------------------------------------------------------
// Kernel 3: scatter — weight = exp(rowmax - colmax[assign]); accum += w*value
// grid: N/64 = 2048 blocks x 256 threads (one thread per feature dim)
// ---------------------------------------------------------------------------
__global__ __launch_bounds__(BLOCK) void scatter_kernel(
    const float* __restrict__ value,
    const float* __restrict__ rowmax,
    const int*   __restrict__ assign,
    const float* __restrict__ colmax,
    float* __restrict__ accum) {
  const int tid = threadIdx.x;
  const int q0  = blockIdx.x * 64;
  for (int j = 0; j < 64; ++j) {
    const int q = q0 + j;                       // uniform -> scalar loads
    if (j + 4 < 64)
      __builtin_prefetch(&value[(size_t)(q + 4) * DD + tid], 0, 0);
    const int   a = assign[q];
    const float w = expf(rowmax[q] - colmax[a]);
    atomicAdd(&accum[a * DD + tid], w * value[(size_t)q * DD + tid]);
  }
}

// ---------------------------------------------------------------------------
// Kernel 4: finalize — out = l2_normalize(accum + keys) per row
// grid: M = 512 blocks x 256 threads (one thread per dim)
// ---------------------------------------------------------------------------
__global__ __launch_bounds__(BLOCK) void finalize_kernel(
    const float* __restrict__ keys,
    const float* __restrict__ accum,
    float* __restrict__ out) {
  __shared__ float red[8];
  const int row = blockIdx.x;
  const int tid = threadIdx.x;
  float v = accum[row * DD + tid] + keys[row * DD + tid];
  float s = v * v;
#pragma unroll
  for (int m = 16; m >= 1; m >>= 1) s += __shfl_xor(s, m, 32);
  if ((tid & 31) == 0) red[tid >> 5] = s;
  __syncthreads();
  float tot = 0.0f;
#pragma unroll
  for (int i = 0; i < 8; ++i) tot += red[i];
  float n = sqrtf(tot);
  out[row * DD + tid] = v / fmaxf(n, 1e-12f);
}

// ---------------------------------------------------------------------------
// Launch
// ---------------------------------------------------------------------------
extern "C" void kernel_launch(void* const* d_in, const int* in_sizes, int n_in,
                              void* d_out, int out_size, void* d_ws, size_t ws_size,
                              hipStream_t stream) {
  const float* keys  = (const float*)d_in[0];   // [M, D]
  const float* query = (const float*)d_in[1];   // [N, D]
  const float* value = (const float*)d_in[2];   // [N, D]
  float* out = (float*)d_out;                   // [M, D]

  // workspace layout (bytes), ~2.05 MB total
  char* ws = (char*)d_ws;
  unsigned short* khi    = (unsigned short*)(ws);                      // 256 KB
  unsigned short* klo    = (unsigned short*)(ws + 262144);             // 256 KB
  float*          colmax = (float*)(ws + 524288);                      //   2 KB
  float*          rowmax = (float*)(ws + 526336);                      // 512 KB
  int*            assign = (int*)  (ws + 1050624);                     // 512 KB
  float*          accum  = (float*)(ws + 1574912);                     // 512 KB

  prep_kernel<<<(MM * DD) / BLOCK, BLOCK, 0, stream>>>(keys, khi, klo, accum, colmax);

  const size_t lds_bytes = (size_t)2 * QTILE * ROWQ * sizeof(unsigned short)
                         + (size_t)MM * sizeof(float);                 // 137216 B
  score_kernel<<<NN / QTILE, BLOCK, lds_bytes, stream>>>(
      query, khi, klo, colmax, rowmax, assign);

  scatter_kernel<<<NN / 64, BLOCK, 0, stream>>>(value, rowmax, assign, colmax, accum);

  finalize_kernel<<<MM, BLOCK, 0, stream>>>(keys, accum, out);
}
